// GNN_50508815401073
// MI455X (gfx1250) — matrix operand order repaired and checked by reading, not compiled, not run on hardware
//
#include <hip/hip_runtime.h>
#include <hip/hip_bf16.h>

typedef __attribute__((ext_vector_type(16))) _Float16 v16h;
typedef __attribute__((ext_vector_type(8)))  float    v8f;

constexpr int KD   = 128;  // node feature dim
constexpr int KEF  = 32;   // edge feature dim
constexpr int KDF  = 32;   // dist feature dim
constexpr int KIN  = KD + KEF + KDF;  // 192
constexpr int KATT = 128;
constexpr int KG   = 4 * KD;          // 512 LSTM gate width
constexpr int ET   = 32;              // edges per block (2 WMMA row tiles)

// ---------------------------------------------------------------------------
// WMMA helper: D = A(16x32 f16) * B(32x16 f16) + C(16x16 f32)
// ---------------------------------------------------------------------------
__device__ __forceinline__ v8f wmma_f16(v16h a, v16h b, v8f c) {
    return __builtin_amdgcn_wmma_f32_16x16x32_f16(
        /*neg_a=*/false, a, /*neg_b=*/false, b,
        /*c_mod=*/(short)0, c, /*reuse_a=*/false, /*reuse_b=*/false);
}

// ---------------------------------------------------------------------------
// A fragment: 16x32 f16 tile, row-major source (LDS), stride ldm elements.
// Lanes 0-15 row M=lane hold K=0..15; lanes 16-31 K=16..31.
// Each half is a contiguous 16-byte chunk -> 2x ds_load_b128.
// ---------------------------------------------------------------------------
__device__ __forceinline__ v16h load_frag_a_lds(const _Float16* base, int ldm, int lane) {
    const int row  = lane & 15;
    const int koff = (lane & 16) ? 16 : 0;
    const _Float16* p = base + row * ldm;
    v16h a;
#pragma unroll
    for (int i = 0; i < 8; ++i) a[i]     = p[koff + i];
#pragma unroll
    for (int i = 0; i < 8; ++i) a[8 + i] = p[koff + 8 + i];
    return a;
}

// ---------------------------------------------------------------------------
// B fragment from pre-packed fragment-order buffer: each (tile, lane) owns 16
// contiguous halfs (32 B, 32 B-aligned) -> single v16h load (2x b128).
// tile = tileN * numKc + kc.
// ---------------------------------------------------------------------------
__device__ __forceinline__ v16h load_frag_b_packed(const _Float16* __restrict__ packed,
                                                   int tile, int lane) {
    return *(const v16h*)(packed + ((size_t)tile * 32 + lane) * 16);
}

__device__ __forceinline__ float sigmoidf_fast(float x) {
    return 1.0f / (1.0f + __expf(-x));
}

// ---------------------------------------------------------------------------
// Weight repack kernels (run once per launch; tiny).
// Fragment order: dst[((tileN*numKc + kc)*32 + lane)*16 + j], where
//   col = lane&15, koff = (lane&16)?16:0, k = kc*32+koff+j, n = tileN*16+col.
// ---------------------------------------------------------------------------
__global__ void pack_kxn_kernel(const float* __restrict__ src, _Float16* __restrict__ dst,
                                int K, int Nn) {   // src is K x Nn row-major
    const int idx = blockIdx.x * 256 + threadIdx.x;
    if (idx >= K * Nn) return;
    const int j    = idx & 15;
    const int lane = (idx >> 4) & 31;
    const int tile = idx >> 9;
    const int numKc = K / 32;
    const int tn = tile / numKc;
    const int kc = tile - tn * numKc;
    const int col  = lane & 15;
    const int koff = (lane & 16) ? 16 : 0;
    const int k = kc * 32 + koff + j;
    const int n = tn * 16 + col;
    dst[idx] = (_Float16)src[(size_t)k * Nn + n];
}

__global__ void pack_nxk_kernel(const float* __restrict__ src, _Float16* __restrict__ dst,
                                int K, int Nn) {   // src is Nn x K row-major (W, used as W^T)
    const int idx = blockIdx.x * 256 + threadIdx.x;
    if (idx >= K * Nn) return;
    const int j    = idx & 15;
    const int lane = (idx >> 4) & 31;
    const int tile = idx >> 9;
    const int numKc = K / 32;
    const int tn = tile / numKc;
    const int kc = tile - tn * numKc;
    const int col  = lane & 15;
    const int koff = (lane & 16) ? 16 : 0;
    const int k = kc * 32 + koff + j;
    const int n = tn * 16 + col;
    dst[idx] = (_Float16)src[(size_t)n * K + k];
}

// ---------------------------------------------------------------------------
// Edge kernel: one block = 32 edges (2 WMMA row tiles -> each B fragment
// feeds 2 WMMAs). Fused gather -> concat -> MLP(msg) & MLP(att) -> sigmoid
// gating -> atomic segment-sum. 8 waves; wave w owns columns [16w, 16w+16).
// ---------------------------------------------------------------------------
__global__ void __launch_bounds__(256)
edge_kernel(const float* __restrict__ state,
            const int*   __restrict__ edge,
            const float* __restrict__ edge_feat,
            const float* __restrict__ dist_feat,
            const _Float16* __restrict__ p_msg_w0, const float* __restrict__ msg_b0,
            const _Float16* __restrict__ p_msg_w1, const float* __restrict__ msg_b1,
            const _Float16* __restrict__ p_att_w0, const float* __restrict__ att_b0,
            const _Float16* __restrict__ p_att_w1, const float* __restrict__ att_b1,
            float* __restrict__ out_msg,
            int M, int relu_state)
{
    __shared__ _Float16 s_in[ET * KIN];    // 12 KB: edge_input tile (32 x 192)
    __shared__ _Float16 s_hid[ET * KD];    //  8 KB: relu(msg hidden)
    __shared__ _Float16 s_ahid[ET * KATT]; //  8 KB: relu(att hidden)
    __shared__ int s_dst[ET];

    const int e0   = blockIdx.x * ET;
    const int tid  = threadIdx.x;
    const int lane = tid & 31;
    const int wave = tid >> 5;

    // ---- Stage 1: build 32x192 edge_input tile (gathers hit L2) ----
    for (int idx = tid; idx < ET * KIN; idx += 256) {
        const int r = idx / KIN;
        const int ccol = idx - r * KIN;
        const int e = e0 + r;
        float v = 0.0f;
        if (e < M) {
            if (ccol < KD) {
                const int s = edge[2 * e];
                const int d = edge[2 * e + 1];
                float a = state[(size_t)s * KD + ccol];
                float b = state[(size_t)d * KD + ccol];
                if (relu_state) { a = a > 0.f ? a : 0.f; b = b > 0.f ? b : 0.f; }
                v = a - b;
            } else if (ccol < KD + KEF) {
                v = edge_feat[(size_t)e * KEF + (ccol - KD)];
            } else {
                v = dist_feat[(size_t)e * KDF + (ccol - KD - KEF)];
            }
        }
        s_in[r * KIN + ccol] = (_Float16)v;
    }
    if (tid < ET) {
        const int e = e0 + tid;
        s_dst[tid] = (e < M) ? edge[2 * e + 1] : -1;
    }
    __syncthreads();

    const int n0 = wave * 16;   // this wave's output column base (0..112)

    // ---- Stage 2: GEMM1 (K = 192) for msg & att hidden ----
    // B fragments shared across both row tiles; A fragments shared msg/att.
    v8f accm[2] = {}, acca[2] = {};
#pragma unroll
    for (int kc = 0; kc < KIN / 32; ++kc) {
        const v16h bm = load_frag_b_packed(p_msg_w0, wave * (KIN / 32) + kc, lane);
        const v16h ba = load_frag_b_packed(p_att_w0, wave * (KIN / 32) + kc, lane);
#pragma unroll
        for (int rt = 0; rt < 2; ++rt) {
            const v16h a = load_frag_a_lds(s_in + rt * 16 * KIN + kc * 32, KIN, lane);
            accm[rt] = wmma_f16(a, bm, accm[rt]);
            acca[rt] = wmma_f16(a, ba, acca[rt]);
        }
    }
    {
        const int col   = lane & 15;
        const int rbase = (lane & 16) ? 8 : 0;
        const float bm = msg_b0[n0 + col];
        const float ba = att_b0[n0 + col];
#pragma unroll
        for (int rt = 0; rt < 2; ++rt) {
#pragma unroll
            for (int r = 0; r < 8; ++r) {
                float hm = accm[rt][r] + bm; hm = hm > 0.f ? hm : 0.f;
                float ha = acca[rt][r] + ba; ha = ha > 0.f ? ha : 0.f;
                const int row = rt * 16 + rbase + r;
                s_hid [row * KD   + n0 + col] = (_Float16)hm;
                s_ahid[row * KATT + n0 + col] = (_Float16)ha;
            }
        }
    }
    __syncthreads();

    // ---- Stage 3: GEMM2 (K = 128), gate, atomic scatter-add ----
    v8f accm2[2] = {}, acca2[2] = {};
#pragma unroll
    for (int kc = 0; kc < KD / 32; ++kc) {
        const v16h bm = load_frag_b_packed(p_msg_w1, wave * (KD / 32) + kc, lane);
        const v16h ba = load_frag_b_packed(p_att_w1, wave * (KD / 32) + kc, lane);
#pragma unroll
        for (int rt = 0; rt < 2; ++rt) {
            const v16h am = load_frag_a_lds(s_hid  + rt * 16 * KD   + kc * 32, KD,   lane);
            const v16h aa = load_frag_a_lds(s_ahid + rt * 16 * KATT + kc * 32, KATT, lane);
            accm2[rt] = wmma_f16(am, bm, accm2[rt]);
            acca2[rt] = wmma_f16(aa, ba, acca2[rt]);
        }
    }
    {
        const int col   = lane & 15;
        const int rbase = (lane & 16) ? 8 : 0;
        const float bm = msg_b1[n0 + col];
        const float ba = att_b1[n0 + col];
#pragma unroll
        for (int rt = 0; rt < 2; ++rt) {
#pragma unroll
            for (int r = 0; r < 8; ++r) {
                const int dn = s_dst[rt * 16 + rbase + r];
                if (dn >= 0) {
                    const float m  = accm2[rt][r] + bm;
                    const float sg = sigmoidf_fast(acca2[rt][r] + ba);
                    atomicAdd(&out_msg[(size_t)dn * KD + n0 + col], m * sg);
                }
            }
        }
    }
}

// ---------------------------------------------------------------------------
// LSTM kernel: one block = 16 nodes. gates(16x512) = x@Wih^T + h@Whh^T + b,
// then elementwise cell update. In-place h/c update; optional mirror to out.
// ---------------------------------------------------------------------------
__global__ void __launch_bounds__(256)
lstm_kernel(const float* __restrict__ x,     // state_msg, N x 128
            float* __restrict__ h,           // state (in/out)
            float* __restrict__ c,           // cell  (in/out)
            const _Float16* __restrict__ p_wih,   // packed 128x512 (W^T)
            const _Float16* __restrict__ p_whh,   // packed 128x512 (W^T)
            const float* __restrict__ bih,   // 512
            const float* __restrict__ bhh,   // 512
            float* __restrict__ h_mirror,    // final-layer output or nullptr
            int N, int relu_h)
{
    __shared__ _Float16 s_x[16 * KD];   // 4 KB
    __shared__ _Float16 s_h[16 * KD];   // 4 KB
    __shared__ float    s_g[16 * KG];   // 32 KB gates

    const int node0 = blockIdx.x * 16;
    const int tid  = threadIdx.x;
    const int lane = tid & 31;
    const int wave = tid >> 5;

    for (int idx = tid; idx < 16 * KD; idx += 256) {
        const int r = idx >> 7;
        const int col = idx & (KD - 1);
        const int node = node0 + r;
        float xv = 0.f, hv = 0.f;
        if (node < N) {
            xv = x[(size_t)node * KD + col];
            hv = h[(size_t)node * KD + col];
            if (relu_h) hv = hv > 0.f ? hv : 0.f;
        }
        s_x[idx] = (_Float16)xv;
        s_h[idx] = (_Float16)hv;
    }
    __syncthreads();

    // Each wave computes 4 gate-column tiles of 16 (8 waves x 64 = 512 cols).
#pragma unroll
    for (int t = 0; t < 4; ++t) {
        const int gtile = wave * 4 + t;   // N-tile index, 0..31
        const int gn0 = gtile * 16;
        v8f acc = {};
#pragma unroll
        for (int kc = 0; kc < KD / 32; ++kc) {
            const v16h ax = load_frag_a_lds(s_x + kc * 32, KD, lane);
            const v16h ah = load_frag_a_lds(s_h + kc * 32, KD, lane);
            const v16h bx = load_frag_b_packed(p_wih, gtile * (KD / 32) + kc, lane);
            const v16h bh = load_frag_b_packed(p_whh, gtile * (KD / 32) + kc, lane);
            acc = wmma_f16(ax, bx, acc);
            acc = wmma_f16(ah, bh, acc);
        }
        const int col   = lane & 15;
        const int rbase = (lane & 16) ? 8 : 0;
        const float bb = bih[gn0 + col] + bhh[gn0 + col];
#pragma unroll
        for (int r = 0; r < 8; ++r)
            s_g[(rbase + r) * KG + gn0 + col] = acc[r] + bb;
    }
    __syncthreads();

    // Elementwise LSTM cell update.
    for (int idx = tid; idx < 16 * KD; idx += 256) {
        const int r = idx >> 7;
        const int d = idx & (KD - 1);
        const int node = node0 + r;
        if (node < N) {
            const float gi = s_g[r * KG + d];
            const float gf = s_g[r * KG + KD + d];
            const float gg = s_g[r * KG + 2 * KD + d];
            const float go = s_g[r * KG + 3 * KD + d];
            const float cv = c[(size_t)node * KD + d];
            const float c2 = sigmoidf_fast(gf) * cv + sigmoidf_fast(gi) * tanhf(gg);
            const float h2 = sigmoidf_fast(go) * tanhf(c2);
            c[(size_t)node * KD + d] = c2;
            h[(size_t)node * KD + d] = h2;
            if (h_mirror) h_mirror[(size_t)node * KD + d] = h2;
        }
    }
}

// ---------------------------------------------------------------------------
// Host-side orchestration (graph-capture safe: async memcpy/memset only).
// ---------------------------------------------------------------------------
extern "C" void kernel_launch(void* const* d_in, const int* in_sizes, int n_in,
                              void* d_out, int out_size, void* d_ws, size_t ws_size,
                              hipStream_t stream) {
    const float* node_feat   = (const float*)d_in[0];
    const float* node_feat_c = (const float*)d_in[1];
    const int*   edge        = (const int*)  d_in[2];
    const float* edge_feat   = (const float*)d_in[3];
    const float* dist_feat   = (const float*)d_in[4];
    const float* msg_w0      = (const float*)d_in[5];
    const float* msg_b0      = (const float*)d_in[6];
    const float* msg_w1      = (const float*)d_in[7];
    const float* msg_b1      = (const float*)d_in[8];
    const float* att_w0      = (const float*)d_in[9];
    const float* att_b0      = (const float*)d_in[10];
    const float* att_w1      = (const float*)d_in[11];
    const float* att_b1      = (const float*)d_in[12];
    const float* lstm_wih    = (const float*)d_in[13];
    const float* lstm_whh    = (const float*)d_in[14];
    const float* lstm_bih    = (const float*)d_in[15];
    const float* lstm_bhh    = (const float*)d_in[16];

    const int N = in_sizes[0] / KD;
    const int M = in_sizes[2] / 2;
    const int L = 2;

    // ---- Workspace layout ----
    float* state  = (float*)d_ws;                 // N x 128
    float* cst    = state + (size_t)N * KD;       // N x 128
    float* msgbuf = cst   + (size_t)N * KD;       // N x 128 (segment-sum accum)
    _Float16* pw  = (_Float16*)(msgbuf + (size_t)N * KD);   // packed weights

    const size_t SZ_W0 = (size_t)KIN * KD;   // 24576 halfs
    const size_t SZ_W1 = (size_t)KD * KD;    // 16384
    const size_t SZ_WG = (size_t)KD * KG;    // 65536
    const size_t PER_LAYER = 2 * SZ_W0 + 2 * SZ_W1 + 2 * SZ_WG;

    const size_t nd_bytes = (size_t)N * KD * sizeof(float);
    hipMemcpyAsync(state, node_feat,   nd_bytes, hipMemcpyDeviceToDevice, stream);
    hipMemcpyAsync(cst,   node_feat_c, nd_bytes, hipMemcpyDeviceToDevice, stream);

    // ---- Pre-pack all weights into fragment-native f16 layout ----
    for (int ii = 0; ii < L; ++ii) {
        _Float16* base = pw + (size_t)ii * PER_LAYER;
        _Float16* pm0 = base;
        _Float16* pa0 = pm0 + SZ_W0;
        _Float16* pm1 = pa0 + SZ_W0;
        _Float16* pa1 = pm1 + SZ_W1;
        _Float16* pih = pa1 + SZ_W1;
        _Float16* phh = pih + SZ_WG;

        pack_kxn_kernel<<<(int)((SZ_W0 + 255) / 256), 256, 0, stream>>>(
            msg_w0 + (size_t)ii * KIN * KD, pm0, KIN, KD);
        pack_kxn_kernel<<<(int)((SZ_W0 + 255) / 256), 256, 0, stream>>>(
            att_w0 + (size_t)ii * KIN * KATT, pa0, KIN, KATT);
        pack_kxn_kernel<<<(int)((SZ_W1 + 255) / 256), 256, 0, stream>>>(
            msg_w1 + (size_t)ii * KD * KD, pm1, KD, KD);
        pack_kxn_kernel<<<(int)((SZ_W1 + 255) / 256), 256, 0, stream>>>(
            att_w1 + (size_t)ii * KATT * KD, pa1, KATT, KD);
        pack_nxk_kernel<<<(int)((SZ_WG + 255) / 256), 256, 0, stream>>>(
            lstm_wih + (size_t)ii * KG * KD, pih, KD, KG);
        pack_nxk_kernel<<<(int)((SZ_WG + 255) / 256), 256, 0, stream>>>(
            lstm_whh + (size_t)ii * KG * KD, phh, KD, KG);
    }

    const int eblocks = (M + ET - 1) / ET;
    const int nblocks = (N + 15) / 16;

    for (int ii = 0; ii < L; ++ii) {
        _Float16* base = pw + (size_t)ii * PER_LAYER;
        _Float16* pm0 = base;
        _Float16* pa0 = pm0 + SZ_W0;
        _Float16* pm1 = pa0 + SZ_W0;
        _Float16* pa1 = pm1 + SZ_W1;
        _Float16* pih = pa1 + SZ_W1;
        _Float16* phh = pih + SZ_WG;

        hipMemsetAsync(msgbuf, 0, nd_bytes, stream);

        edge_kernel<<<eblocks, 256, 0, stream>>>(
            state, edge, edge_feat, dist_feat,
            pm0, msg_b0 + (size_t)ii * KD,
            pm1, msg_b1 + (size_t)ii * KD,
            pa0, att_b0 + (size_t)ii * KATT,
            pa1, att_b1 + (size_t)ii * KD,
            msgbuf, M, /*relu_state=*/ii);

        lstm_kernel<<<nblocks, 256, 0, stream>>>(
            msgbuf, state, cst,
            pih, phh,
            lstm_bih + (size_t)ii * KG, lstm_bhh + (size_t)ii * KG,
            (ii == L - 1) ? (float*)d_out : nullptr, N, /*relu_h=*/ii);
    }
}
